// TorchPaiNN_16819091931692
// MI455X (gfx1250) — compile-verified, matrix operand order checked by reading, not run on hardware
//
#include <hip/hip_runtime.h>
#include <hip/hip_bf16.h>
#include <math.h>

#define NATOMS  10000
#define NEDGES  320000
#define CH      128
#define NBASIS  32
#define NLAYERS 3
#define CUTOFF  5.0f

typedef __attribute__((ext_vector_type(16))) __bf16 v16bf;
typedef __attribute__((ext_vector_type(8)))  float  v8f;

// ---------------- fragment packing (fp32 -> bf16, WMMA lane layout) ----------------
// A [M,K] row-major -> Ap[((mt*Ktiles+kt)*32+lane)*16] : lane's v16bf A-fragment.
__global__ void pack_a_kernel(const float* __restrict__ A, __bf16* __restrict__ Ap,
                              int M, int K) {
  int t = blockIdx.x * blockDim.x + threadIdx.x;
  int Ktiles = K >> 5;
  int total  = (M >> 4) * Ktiles * 32;
  if (t >= total) return;
  int lane = t & 31;
  int kt   = (t >> 5) % Ktiles;
  int mt   = (t >> 5) / Ktiles;
  const float* rowp = A + (size_t)((mt << 4) + (lane & 15)) * K + (kt << 5)
                      + ((lane >> 4) & 1) * 8;
  __bf16* o = Ap + (size_t)t * 16;
#pragma unroll
  for (int i = 0; i < 8; ++i) {
    int kb = ((i & 4) << 2) + ((i & 3) << 1);   // {0,2,4,6,16,18,20,22}
    o[2 * i]     = (__bf16)rowp[kb];
    o[2 * i + 1] = (__bf16)rowp[kb + 1];
  }
}

// B [K,N] row-major -> Bp[((nt*Ktiles+kt)*32+lane)*16] : lane's v16bf B-fragment.
__global__ void pack_b_kernel(const float* __restrict__ B, __bf16* __restrict__ Bp,
                              int K, int Ndim) {
  int t = blockIdx.x * blockDim.x + threadIdx.x;
  int Ktiles = K >> 5;
  int total  = (Ndim >> 4) * Ktiles * 32;
  if (t >= total) return;
  int lane = t & 31;
  int kt   = (t >> 5) % Ktiles;
  int nt   = (t >> 5) / Ktiles;
  int n     = (nt << 4) + (lane & 15);
  int kbase = (kt << 5) + ((lane >> 4) & 1) * 8;
  __bf16* o = Bp + (size_t)t * 16;
#pragma unroll
  for (int i = 0; i < 8; ++i) {
    int k = kbase + ((i & 4) << 2) + ((i & 3) << 1);
    o[2 * i]     = (__bf16)B[(size_t)k * Ndim + n];
    o[2 * i + 1] = (__bf16)B[(size_t)(k + 1) * Ndim + n];
  }
}

// ---------------- packed WMMA GEMM: one wave = 16(M) x 64(N) tile ----------------
// Inner loop per K-step: 1 A-fragment (2x b128) + 4 B-fragments + 4 WMMAs.
// Grid is exact; EXEC all-ones at every WMMA.
template <int ACT, bool HB>
__global__ __launch_bounds__(32)
void wmma_gemm_packed_kernel(const __bf16* __restrict__ Ap, const __bf16* __restrict__ Bp,
                             const float* __restrict__ bias, float* __restrict__ Cmat,
                             int M, int K, int Ndim) {
  int lane = threadIdx.x;
  int n64  = Ndim >> 6;
  int mt   = blockIdx.x / n64;
  int nt64 = blockIdx.x - mt * n64;
  int Ktiles = K >> 5;

  const v16bf* ap  = (const v16bf*)(Ap + ((size_t)(mt * Ktiles) * 32 + lane) * 16);
  const v16bf* bp0 = (const v16bf*)(Bp + ((size_t)((nt64 * 4 + 0) * Ktiles) * 32 + lane) * 16);
  const v16bf* bp1 = (const v16bf*)(Bp + ((size_t)((nt64 * 4 + 1) * Ktiles) * 32 + lane) * 16);
  const v16bf* bp2 = (const v16bf*)(Bp + ((size_t)((nt64 * 4 + 2) * Ktiles) * 32 + lane) * 16);
  const v16bf* bp3 = (const v16bf*)(Bp + ((size_t)((nt64 * 4 + 3) * Ktiles) * 32 + lane) * 16);

  v8f acc0 = {}, acc1 = {}, acc2 = {}, acc3 = {};
  for (int kt = 0; kt < Ktiles; ++kt) {
    v16bf fa = ap[(size_t)kt * 32];
    v16bf b0 = bp0[(size_t)kt * 32];
    v16bf b1 = bp1[(size_t)kt * 32];
    v16bf b2 = bp2[(size_t)kt * 32];
    v16bf b3 = bp3[(size_t)kt * 32];
    acc0 = __builtin_amdgcn_wmma_f32_16x16x32_bf16(false, fa, false, b0, (short)0, acc0, false, false);
    acc1 = __builtin_amdgcn_wmma_f32_16x16x32_bf16(false, fa, false, b1, (short)0, acc1, false, false);
    acc2 = __builtin_amdgcn_wmma_f32_16x16x32_bf16(false, fa, false, b2, (short)0, acc2, false, false);
    acc3 = __builtin_amdgcn_wmma_f32_16x16x32_bf16(false, fa, false, b3, (short)0, acc3, false, false);
  }

  int nloc = lane & 15;
  int mrow = (mt << 4) + ((lane >> 4) << 3);
  float* crow = Cmat + (size_t)mrow * Ndim + (nt64 << 6) + nloc;
  v8f accs[4] = {acc0, acc1, acc2, acc3};
#pragma unroll
  for (int s = 0; s < 4; ++s) {
    float bv = HB ? bias[(nt64 << 6) + (s << 4) + nloc] : 0.0f;
#pragma unroll
    for (int j = 0; j < 8; ++j) {
      float x = accs[s][j] + bv;
      if (ACT) x = x * __builtin_amdgcn_rcpf(1.0f + __expf(-x));   // fast silu
      crow[(size_t)j * Ndim + (s << 4)] = x;
    }
  }
}

// ---------------- elementwise / graph kernels ----------------
__global__ void embed_kernel(const int* __restrict__ z, const float* __restrict__ emb,
                             float* __restrict__ scalar) {
  int idx = blockIdx.x * blockDim.x + threadIdx.x;
  if (idx >= NATOMS * CH) return;
  int n = idx >> 7, c = idx & 127;
  scalar[idx] = emb[z[n] * CH + c];
}

__global__ void edge_geom_kernel(const float* __restrict__ pos, const float* __restrict__ shifts,
                                 const int* __restrict__ ei, float* __restrict__ dir,
                                 float* __restrict__ elen, float* __restrict__ gate) {
  int e = blockIdx.x * blockDim.x + threadIdx.x;
  if (e >= NEDGES) return;
  int r = ei[e], c = ei[NEDGES + e];
  float dx = pos[c * 3 + 0] - pos[r * 3 + 0] + shifts[e * 3 + 0];
  float dy = pos[c * 3 + 1] - pos[r * 3 + 1] + shifts[e * 3 + 1];
  float dz = pos[c * 3 + 2] - pos[r * 3 + 2] + shifts[e * 3 + 2];
  float len = sqrtf(dx * dx + dy * dy + dz * dz + 1e-12f);
  float inv = 1.0f / len;
  dir[e * 3 + 0] = dx * inv;
  dir[e * 3 + 1] = dy * inv;
  dir[e * 3 + 2] = dz * inv;
  elen[e] = len;
  float s = len / CUTOFF, g = 0.0f;
  if (s < 1.0f) { float denom = 1.0f - s * s; g = __expf(1.0f - 1.0f / denom); }
  gate[e] = g;
}

// One block (128 threads) per edge: rbf in LDS, fused radial filter, scatter atomics.
__global__ void edge_message_kernel(const int* __restrict__ ei, const float* __restrict__ elen,
                                    const float* __restrict__ gate, const float* __restrict__ dir,
                                    const float* __restrict__ hnode, const float* __restrict__ vec,
                                    const float* __restrict__ fw, const float* __restrict__ fb,
                                    float* __restrict__ dsca, float* __restrict__ dvec) {
  __shared__ float s_rbf[NBASIS];
  int   e = blockIdx.x;
  int   c = threadIdx.x;
  float g = gate[e];
  if (c < NBASIS) {
    const float delta = CUTOFF / (NBASIS - 1);
    const float gamma = 1.0f / (delta * delta);
    float d = elen[e] - c * delta;
    s_rbf[c] = __expf(-gamma * d * d) * g;    // edge_attr = rbf * gate
  }
  __syncthreads();
  if (g == 0.0f) return;                      // uniform across block
  int r = ei[e], cl = ei[NEDGES + e];
  float f1 = fb[c], f2 = fb[CH + c], f3 = fb[2 * CH + c];
#pragma unroll
  for (int k = 0; k < NBASIS; ++k) {
    float rb = s_rbf[k];
    const float* fr = fw + (size_t)k * 3 * CH;
    f1 += rb * fr[c];
    f2 += rb * fr[CH + c];
    f3 += rb * fr[2 * CH + c];
  }
  const float* hn = hnode + (size_t)cl * 3 * CH;
  float w1 = hn[c] * f1 * g;
  float w2 = hn[CH + c] * f2 * g;
  float w3 = hn[2 * CH + c] * f3 * g;
  __hip_atomic_fetch_add(&dsca[(size_t)r * CH + c], w1,
                         __ATOMIC_RELAXED, __HIP_MEMORY_SCOPE_AGENT);
  const float* vc  = vec  + (size_t)cl * 3 * CH;
  float*       dvr = dvec + (size_t)r  * 3 * CH;
#pragma unroll
  for (int d = 0; d < 3; ++d) {
    float m = vc[d * CH + c] * w2 + w3 * dir[e * 3 + d];
    __hip_atomic_fetch_add(&dvr[d * CH + c], m,
                           __ATOMIC_RELAXED, __HIP_MEMORY_SCOPE_AGENT);
  }
}

__global__ void add_inplace_kernel(float* __restrict__ x, const float* __restrict__ d, int n) {
  int i = blockIdx.x * blockDim.x + threadIdx.x;
  if (i < n) x[i] += d[i];
}

__global__ void vnorm_dot_kernel(const float* __restrict__ scalar, const float* __restrict__ uv,
                                 const float* __restrict__ vv, float* __restrict__ cat,
                                 float* __restrict__ dotb) {
  int idx = blockIdx.x * blockDim.x + threadIdx.x;
  if (idx >= NATOMS * CH) return;
  int n = idx >> 7, c = idx & 127;
  const float* u = uv + (size_t)n * 3 * CH + c;
  const float* v = vv + (size_t)n * 3 * CH + c;
  float v0 = v[0], v1 = v[CH], v2 = v[2 * CH];
  float u0 = u[0], u1 = u[CH], u2 = u[2 * CH];
  cat[(size_t)n * 2 * CH + c]      = scalar[idx];
  cat[(size_t)n * 2 * CH + CH + c] = sqrtf(v0 * v0 + v1 * v1 + v2 * v2 + 1e-8f);
  dotb[idx] = u0 * v0 + u1 * v1 + u2 * v2;
}

__global__ void update_apply_kernel(const float* __restrict__ a, const float* __restrict__ dotb,
                                    const float* __restrict__ uv, float* __restrict__ scalar,
                                    float* __restrict__ vec) {
  int idx = blockIdx.x * blockDim.x + threadIdx.x;
  if (idx >= NATOMS * CH) return;
  int n = idx >> 7, c = idx & 127;
  const float* an = a + (size_t)n * 3 * CH;
  float a1 = an[c], a2 = an[CH + c], a3 = an[2 * CH + c];
  scalar[idx] += a1 + a2 * dotb[idx];
  float*       vn = vec + (size_t)n * 3 * CH + c;
  const float* un = uv  + (size_t)n * 3 * CH + c;
  vn[0]      += a3 * un[0];
  vn[CH]     += a3 * un[CH];
  vn[2 * CH] += a3 * un[2 * CH];
}

__global__ void readout_kernel(const float* __restrict__ h, const float* __restrict__ r_w2,
                               const float* __restrict__ r_b2, const int* __restrict__ batch,
                               float* __restrict__ out) {
  int n = blockIdx.x * blockDim.x + threadIdx.x;
  if (n >= NATOMS) return;
  float acc = r_b2[0];
#pragma unroll
  for (int c = 0; c < CH / 2; ++c) acc += h[(size_t)n * (CH / 2) + c] * r_w2[c];
  __hip_atomic_fetch_add(&out[batch[n]], acc, __ATOMIC_RELAXED, __HIP_MEMORY_SCOPE_AGENT);
}

// ---------------- host ----------------
extern "C" void kernel_launch(void* const* d_in, const int* in_sizes, int n_in,
                              void* d_out, int out_size, void* d_ws, size_t ws_size,
                              hipStream_t stream) {
  const int*   z      = (const int*)  d_in[0];
  const float* pos    = (const float*)d_in[1];
  const int*   ei     = (const int*)  d_in[2];
  const float* shifts = (const float*)d_in[3];
  const int*   batch  = (const int*)  d_in[4];
  const float* emb    = (const float*)d_in[5];
  const float* m_w1   = (const float*)d_in[6];
  const float* m_b1   = (const float*)d_in[7];
  const float* m_w2   = (const float*)d_in[8];
  const float* m_b2   = (const float*)d_in[9];
  const float* f_w    = (const float*)d_in[10];
  const float* f_b    = (const float*)d_in[11];
  const float* u_w    = (const float*)d_in[12];
  const float* v_w    = (const float*)d_in[13];
  const float* up_w1  = (const float*)d_in[14];
  const float* up_b1  = (const float*)d_in[15];
  const float* up_w2  = (const float*)d_in[16];
  const float* up_b2  = (const float*)d_in[17];
  const float* r_w1   = (const float*)d_in[18];
  const float* r_b1   = (const float*)d_in[19];
  const float* r_w2   = (const float*)d_in[20];
  const float* r_b2   = (const float*)d_in[21];
  float* out = (float*)d_out;

  const size_t NC = (size_t)NATOMS * CH;
  float* ws     = (float*)d_ws;
  float* scalar = ws;             // NC   (scalar..dvec contiguous for the fused add)
  float* vec    = ws + NC;        // 3*NC
  float* dsca   = ws + 4 * NC;    // NC
  float* dvec   = ws + 5 * NC;    // 3*NC
  float* tmp1   = ws + 8 * NC;    // NC
  float* hnode  = ws + 9 * NC;    // 3*NC (reused as update-MLP output 'a')
  float* cat    = ws + 12 * NC;   // 2*NC
  float* uv     = ws + 14 * NC;   // 3*NC
  float* vv     = ws + 17 * NC;   // 3*NC
  float* dotb   = ws + 20 * NC;   // NC
  float* dir    = ws + 21 * NC;   // 3*E
  float* elen   = dir + 3 * (size_t)NEDGES;   // E
  float* gateb  = elen + NEDGES;              // E
  __bf16* apack = (__bf16*)(gateb + NEDGES);  // 3*NC bf16 (largest A: vec as [3N,CH])
  __bf16* bpack = apack + 3 * NC;             // packed weights

  // packed-weight offsets (bf16 elements), per layer
  const size_t SZ_M1 = CH * CH, SZ_M2 = CH * 3 * CH, SZ_U = CH * CH, SZ_V = CH * CH;
  const size_t SZ_UP1 = 2 * CH * CH, SZ_UP2 = CH * 3 * CH;
  const size_t LYR = SZ_M1 + SZ_M2 + SZ_U + SZ_V + SZ_UP1 + SZ_UP2;
  const size_t OFF_M1 = 0, OFF_M2 = SZ_M1, OFF_U = OFF_M2 + SZ_M2, OFF_V = OFF_U + SZ_U;
  const size_t OFF_UP1 = OFF_V + SZ_V, OFF_UP2 = OFF_UP1 + SZ_UP1;
  const size_t OFF_R1 = NLAYERS * LYR;

  auto packB = [&](const float* B, __bf16* Bp, int K, int Nd) {
    int total = (Nd / 16) * (K / 32) * 32;
    pack_b_kernel<<<(total + 255) / 256, 256, 0, stream>>>(B, Bp, K, Nd);
  };
  auto packA = [&](const float* A, int M, int K) {
    int total = (M / 16) * (K / 32) * 32;
    pack_a_kernel<<<(total + 255) / 256, 256, 0, stream>>>(A, apack, M, K);
  };
  auto gemm = [&](const __bf16* Bp, const float* bias, float* Cm,
                  int M, int K, int Nd, int act) {
    dim3 grid((M / 16) * (Nd / 64)), block(32);     // exact coverage, 1 wave/block
    if (act)
      wmma_gemm_packed_kernel<1, true><<<grid, block, 0, stream>>>(apack, Bp, bias, Cm, M, K, Nd);
    else if (bias)
      wmma_gemm_packed_kernel<0, true><<<grid, block, 0, stream>>>(apack, Bp, bias, Cm, M, K, Nd);
    else
      wmma_gemm_packed_kernel<0, false><<<grid, block, 0, stream>>>(apack, Bp, nullptr, Cm, M, K, Nd);
  };

  // pack all weights once (tiny; reused 625-1875x by the GEMMs)
  for (int l = 0; l < NLAYERS; ++l) {
    packB(m_w1  + (size_t)l * SZ_M1,  bpack + l * LYR + OFF_M1,  CH,     CH);
    packB(m_w2  + (size_t)l * SZ_M2,  bpack + l * LYR + OFF_M2,  CH,     3 * CH);
    packB(u_w   + (size_t)l * SZ_U,   bpack + l * LYR + OFF_U,   CH,     CH);
    packB(v_w   + (size_t)l * SZ_V,   bpack + l * LYR + OFF_V,   CH,     CH);
    packB(up_w1 + (size_t)l * SZ_UP1, bpack + l * LYR + OFF_UP1, 2 * CH, CH);
    packB(up_w2 + (size_t)l * SZ_UP2, bpack + l * LYR + OFF_UP2, CH,     3 * CH);
  }
  packB(r_w1, bpack + OFF_R1, CH, CH / 2);

  hipMemsetAsync(vec, 0, 3 * NC * sizeof(float), stream);
  embed_kernel<<<(NATOMS * CH + 255) / 256, 256, 0, stream>>>(z, emb, scalar);
  edge_geom_kernel<<<(NEDGES + 255) / 256, 256, 0, stream>>>(pos, shifts, ei, dir, elen, gateb);

  for (int l = 0; l < NLAYERS; ++l) {
    __bf16* wb = bpack + (size_t)l * LYR;
    // per-NODE message MLP (32x fewer FLOPs than per-edge), gathered in edge kernel
    packA(scalar, NATOMS, CH);
    gemm(wb + OFF_M1, m_b1 + l * CH, tmp1, NATOMS, CH, CH, 1);
    packA(tmp1, NATOMS, CH);
    gemm(wb + OFF_M2, m_b2 + l * 3 * CH, hnode, NATOMS, CH, 3 * CH, 0);

    hipMemsetAsync(dsca, 0, 4 * NC * sizeof(float), stream);   // dsca + dvec
    edge_message_kernel<<<NEDGES, CH, 0, stream>>>(ei, elen, gateb, dir, hnode, vec,
                                                   f_w + (size_t)l * NBASIS * 3 * CH,
                                                   f_b + l * 3 * CH, dsca, dvec);
    add_inplace_kernel<<<(int)((4 * NC + 255) / 256), 256, 0, stream>>>(scalar, dsca,
                                                                        (int)(4 * NC));
    // update block: one A-pack of vector feeds both uv and vv GEMMs
    packA(vec, 3 * NATOMS, CH);
    gemm(wb + OFF_U, nullptr, uv, 3 * NATOMS, CH, CH, 0);
    gemm(wb + OFF_V, nullptr, vv, 3 * NATOMS, CH, CH, 0);
    vnorm_dot_kernel<<<(NATOMS * CH + 255) / 256, 256, 0, stream>>>(scalar, uv, vv, cat, dotb);
    packA(cat, NATOMS, 2 * CH);
    gemm(wb + OFF_UP1, up_b1 + l * CH, tmp1, NATOMS, 2 * CH, CH, 1);
    packA(tmp1, NATOMS, CH);
    gemm(wb + OFF_UP2, up_b2 + l * 3 * CH, hnode, NATOMS, CH, 3 * CH, 0);
    update_apply_kernel<<<(NATOMS * CH + 255) / 256, 256, 0, stream>>>(hnode, dotb, uv,
                                                                      scalar, vec);
  }

  // readout: silu(scalar @ r_w1 + r_b1) [N,64] via WMMA, then 64-dot + batch atomic
  packA(scalar, NATOMS, CH);
  gemm(bpack + OFF_R1, r_b1, tmp1, NATOMS, CH, CH / 2, 1);
  hipMemsetAsync(out, 0, (size_t)out_size * sizeof(float), stream);
  readout_kernel<<<(NATOMS + 255) / 256, 256, 0, stream>>>(tmp1, r_w2, r_b2, batch, out);
}
